// HardLSTM_12876311954119
// MI455X (gfx1250) — matrix operand order repaired and checked by reading, not compiled, not run on hardware
//
#include <hip/hip_runtime.h>

// ---------------------------------------------------------------------------
// Bidirectional hard-LSTM for MI455X (gfx1250, wave32, WMMA 16x16x32 bf16).
// T=500, B=32, IN=512, H=1024, gates=4H=4096.
// Phase 1: blocked WMMA GEMM for gx = x@w_ih^T + biases (both directions).
// Phase 2: ONE persistent kernel runs all 500 recurrence steps with an
//          internal device-scope barrier; c-state lives in VGPRs.
// Wave->tile maps are chosen so waves sharing a workgroup reuse each other's
// operand panels through the WGP cache (halves L2 traffic in both phases).
// ---------------------------------------------------------------------------

typedef __attribute__((ext_vector_type(16))) __bf16 v16bf;
typedef __attribute__((ext_vector_type(8)))  float  v8f;

#define T_STEPS 500
#define BATCH   32
#define IN_DIM  512
#define H_DIM   1024
#define G_DIM   4096              // 4*H
#define ROWS    (T_STEPS*BATCH)   // 16000
#define STEP_BLOCKS 64            // persistent kernel grid

__device__ __forceinline__ float hsig(float x) {
    return fminf(fmaxf(0.2f * x + 0.5f, 0.0f), 1.0f);
}
__device__ __forceinline__ float htanh(float x) {
    return fminf(fmaxf(x, -1.0f), 1.0f);
}

// ---------------------------------------------------------------------------
// Pack fp32 row-major [M,K] into bf16 WMMA A-fragments:
// dst [M/16][K/32][lane(32)][elem(16)];  lane = m%16 + 16*((k%16)/8),
//                                        elem = (k/16)*8 + (k%8).
// ---------------------------------------------------------------------------
__global__ void pack_a_bf16(const float* __restrict__ src, __bf16* __restrict__ dst,
                            int M, int K) {
    int id = blockIdx.x * blockDim.x + threadIdx.x;
    if (id >= M * K) return;
    int e    = id & 15;
    int lane = (id >> 4) & 31;
    int tile = id >> 9;
    int Kt = K >> 5;
    int kt = tile % Kt;
    int mt = tile / Kt;
    int m = mt * 16 + (lane & 15);
    int k = kt * 32 + ((e >> 3) << 4) + ((lane >> 4) << 3) + (e & 7);
    dst[id] = (__bf16)src[(size_t)m * K + k];
}

// ---------------------------------------------------------------------------
// Pack fp32 row-major weights w[N,K] into bf16 WMMA B-fragments (B = w^T):
// dst [N/16][K/32][lane(32)][elem(16)];  lane = n%16 + 16*((k%16)/8),
//                                        elem = (k%32/16)*8 + (k%8).
// ---------------------------------------------------------------------------
__global__ void pack_b_bf16(const float* __restrict__ w, __bf16* __restrict__ dst,
                            int N, int K) {
    int id = blockIdx.x * blockDim.x + threadIdx.x;
    if (id >= N * K) return;
    int e    = id & 15;
    int lane = (id >> 4) & 31;
    int tile = id >> 9;
    int Kt = K >> 5;
    int kt = tile % Kt;
    int nt = tile / Kt;
    int n = nt * 16 + (lane & 15);
    int k = kt * 32 + ((e >> 3) << 4) + ((lane >> 4) << 3) + (e & 7);
    dst[id] = (__bf16)w[(size_t)n * K + k];
}

__global__ void zero_bytes(char* p, size_t n) {
    size_t id = (size_t)blockIdx.x * blockDim.x + threadIdx.x;
    if (id < n) p[id] = 0;
}

// ---------------------------------------------------------------------------
// Phase 1: gx[dir][t*B+b][g] = sum_i x[t,b,i]*w_ih[g,i] + b_ih[g] + b_hh[g].
// One wave -> 1 Mtile x 4 Ntiles (64 WMMA, 1 A-load per 4 WMMA).
// Block = 2 Mtiles x 2 Ngroups: every A panel and every B panel is used by
// two waves of the same WGP -> WGP$ catches the second use.
// Grid: 2 dirs * 500 Mpairs * 32 Ngroup-pairs = 32000 blocks, 128 thr/block.
// ---------------------------------------------------------------------------
__global__ __launch_bounds__(128) void gx_gemm(
        const __bf16* __restrict__ xfrag,    // [1000][16][32][16]
        const __bf16* __restrict__ wihfrag,  // [2][256][16][32][16]
        const float* __restrict__ b_ih_f, const float* __restrict__ b_hh_f,
        const float* __restrict__ b_ih_r, const float* __restrict__ b_hh_r,
        float* __restrict__ gx)              // [2][16000][4096]
{
    const int lane = threadIdx.x & 31;
    const int widx = threadIdx.x >> 5;            // 0..3 in block
    const int blk  = blockIdx.x;                  // 0..31999
    const int ntg  = ((blk & 31) << 1) | (widx >> 1);   // group of 4 N-tiles
    const int mt   = (((blk >> 5) % 500) << 1) | (widx & 1);
    const int dir  = blk / 16000;
    const int ncol = lane & 15;
    const int half = lane >> 4;

    v8f acc[4];
#pragma unroll
    for (int q = 0; q < 4; ++q) {
        const int g = (ntg * 4 + q) * 16 + ncol;
        const float bias = dir ? (b_ih_r[g] + b_hh_r[g]) : (b_ih_f[g] + b_hh_f[g]);
#pragma unroll
        for (int r = 0; r < 8; ++r) acc[q][r] = bias;
    }

    const __bf16* ap = xfrag + (((size_t)mt * 16) * 32 + lane) * 16;
    const __bf16* bp = wihfrag + ((((size_t)dir * 256 + ntg * 4) * 16) * 32 + lane) * 16;

#pragma unroll 2
    for (int kt = 0; kt < 16; ++kt) {
        v16bf a = *(const v16bf*)(ap + (size_t)kt * 512);
#pragma unroll
        for (int q = 0; q < 4; ++q) {
            v16bf b = *(const v16bf*)(bp + (size_t)q * 8192 + (size_t)kt * 512);
            acc[q] = __builtin_amdgcn_wmma_f32_16x16x32_bf16(
                    false, a, false, b, (short)0, acc[q], false, false);
        }
    }

    float* gbase = gx + ((size_t)dir * ROWS + (size_t)mt * 16) * G_DIM;
#pragma unroll
    for (int q = 0; q < 4; ++q)
#pragma unroll
        for (int r = 0; r < 8; ++r)
            gbase[(size_t)(r + half * 8) * G_DIM + (ntg * 4 + q) * 16 + ncol] = acc[q][r];
}

// ---------------------------------------------------------------------------
// Device-scope generation barrier: counter is monotonic, step s completes
// when it reaches (s+1)*gridDim.x.  Release via __threadfence (all threads),
// acquire via agent-scope atomic load + __threadfence after the spin.
// ---------------------------------------------------------------------------
__device__ __forceinline__ void grid_barrier(unsigned* bar, int s) {
    __threadfence();
    __syncthreads();
    if (threadIdx.x == 0) {
        __hip_atomic_fetch_add(bar, 1u, __ATOMIC_ACQ_REL, __HIP_MEMORY_SCOPE_AGENT);
        const unsigned target = (unsigned)(s + 1) * STEP_BLOCKS;
        while (__hip_atomic_load(bar, __ATOMIC_ACQUIRE, __HIP_MEMORY_SCOPE_AGENT) < target)
            __builtin_amdgcn_s_sleep(2);
    }
    __syncthreads();
    __threadfence();
}

// ---------------------------------------------------------------------------
// Persistent recurrence kernel: 256 waves (dir, mt=batch tile, jt=h-col tile),
// each wave owns its tile for all 500 steps -> c-state stays in VGPRs.
// Block = {mt0/mt1} x {jt, jt+1} of one direction: every w_hh B panel and
// every h A panel is consumed by two waves in the same WGP (WGP$ reuse).
// Per step: 4 gate tiles, K=1024 -> 32x4 WMMA, C seeded from gx[t].
// h double-buffered in A-fragment layout; hy/cy written at the last step.
// ---------------------------------------------------------------------------
__global__ __launch_bounds__(128) void lstm_persistent(
        const float*  __restrict__ gx,       // [2][16000][4096]
        const __bf16* __restrict__ whhfrag,  // [2][256][32][32][16]
        __bf16* __restrict__ hfragA,         // [2][2][32][32][16]
        __bf16* __restrict__ hfragB,
        float* __restrict__ out,             // [500][32][2048]
        float* __restrict__ hy,              // [2][32][1024]
        float* __restrict__ cy,              // [2][32][1024]
        unsigned* __restrict__ bar)
{
    const int lane = threadIdx.x & 31;
    const int wave = blockIdx.x * 4 + (threadIdx.x >> 5);
    const int dir = wave >> 7;
    const int mt  = wave & 1;                // batch tile: pairs share a block
    const int jt  = (wave >> 1) & 63;        // h-column tile
    const int ncol = lane & 15;
    const int half = lane >> 4;

    // h repack constants (A-fragment layout of next step's h)
    const int j    = jt * 16 + ncol;       // h column 0..1023
    const int kt2  = j >> 5;
    const int e    = ((jt & 1) << 3) + (ncol & 7);
    const int lsel = ((ncol >> 3) & 1) << 4;

    // per-gate B fragment base pointers (weights stay hot in L2/WGP$)
    const __bf16* bp[4];
#pragma unroll
    for (int g = 0; g < 4; ++g)
        bp[g] = whhfrag + ((((size_t)dir * 256 + g * 64 + jt) * 32) * 32 + lane) * 16;
    const size_t a_off = (((size_t)(dir * 2 + mt) * 32) * 32 + lane) * 16;

    v8f cacc;                               // cell state, resident in VGPRs
#pragma unroll
    for (int r = 0; r < 8; ++r) cacc[r] = 0.0f;

    for (int s = 0; s < T_STEPS; ++s) {
        const __bf16* hr = (s & 1) ? hfragB : hfragA;
        __bf16*       hw = (s & 1) ? hfragA : hfragB;
        const int t_io = dir ? (T_STEPS - 1 - s) : s;

        // Seed accumulators from gx tile (C layout: M = r + 8*half, N = ncol).
        const float* gp = gx + ((size_t)dir * ROWS + (size_t)t_io * BATCH) * G_DIM;
        v8f acc[4];
#pragma unroll
        for (int g = 0; g < 4; ++g) {
            const int gcol = g * H_DIM + j;
#pragma unroll
            for (int r = 0; r < 8; ++r)
                acc[g][r] = gp[(size_t)(mt * 16 + r + half * 8) * G_DIM + gcol];
        }

        // Prefetch next step's gx tile behind the WMMA work.
        if (s + 1 < T_STEPS) {
            const int t_n = dir ? (T_STEPS - 2 - s) : (s + 1);
            const float* gq = gx + ((size_t)dir * ROWS + (size_t)t_n * BATCH) * G_DIM
                              + (size_t)(mt * 16 + half * 8) * G_DIM + j;
#pragma unroll
            for (int g = 0; g < 4; ++g)
                __builtin_prefetch(gq + g * H_DIM, 0, 1);
        }

        const __bf16* ap = hr + a_off;
#pragma unroll 2
        for (int kt = 0; kt < 32; ++kt) {
            v16bf a = *(const v16bf*)(ap + (size_t)kt * 512);
#pragma unroll
            for (int g = 0; g < 4; ++g) {
                v16bf b = *(const v16bf*)(bp[g] + (size_t)kt * 512);
                acc[g] = __builtin_amdgcn_wmma_f32_16x16x32_bf16(
                        false, a, false, b, (short)0, acc[g], false, false);
            }
        }

        // Cell update + output + repack h into A-fragment layout.
#pragma unroll
        for (int r = 0; r < 8; ++r) {
            const int brow = mt * 16 + r + half * 8;   // batch index 0..31
            float iv = hsig(acc[0][r]);
            float fv = hsig(acc[1][r]);
            float cg = htanh(acc[2][r]);
            float ov = hsig(acc[3][r]);
            float c = fv * cacc[r] + iv * cg;
            cacc[r] = c;
            float h = ov * htanh(c);
            out[((size_t)t_io * BATCH + brow) * (2 * H_DIM) + dir * H_DIM + j] = h;
            const int mt2 = brow >> 4;
            const int mloc = brow & 15;
            hw[(((size_t)(dir * 2 + mt2) * 32 + kt2) * 32 + (mloc + lsel)) * 16 + e]
                = (__bf16)h;
            if (s == T_STEPS - 1) {
                hy[((size_t)dir * BATCH + brow) * H_DIM + j] = h;
                cy[((size_t)dir * BATCH + brow) * H_DIM + j] = c;
            }
        }

        if (s != T_STEPS - 1) grid_barrier(bar, s);
    }
}

extern "C" void kernel_launch(void* const* d_in, const int* in_sizes, int n_in,
                              void* d_out, int out_size, void* d_ws, size_t ws_size,
                              hipStream_t stream) {
    const float* x      = (const float*)d_in[0];
    const float* w_ih_f = (const float*)d_in[1];
    const float* w_hh_f = (const float*)d_in[2];
    const float* b_ih_f = (const float*)d_in[3];
    const float* b_hh_f = (const float*)d_in[4];
    const float* w_ih_r = (const float*)d_in[5];
    const float* w_hh_r = (const float*)d_in[6];
    const float* b_ih_r = (const float*)d_in[7];
    const float* b_hh_r = (const float*)d_in[8];

    // ---- workspace carve-up -------------------------------------------------
    char* w = (char*)d_ws;
    float* gx = (float*)w;            w += (size_t)2 * ROWS * G_DIM * 4;        // 524 MB
    __bf16* xfrag = (__bf16*)w;       w += (size_t)ROWS * IN_DIM * 2;           // 16.4 MB
    __bf16* wihfrag = (__bf16*)w;     w += (size_t)2 * G_DIM * IN_DIM * 2;      // 8.4 MB
    __bf16* whhfrag = (__bf16*)w;     w += (size_t)2 * G_DIM * H_DIM * 2;       // 16.8 MB
    __bf16* hfragA = (__bf16*)w;      w += (size_t)2 * BATCH * H_DIM * 2;       // 128 KB
    __bf16* hfragB = (__bf16*)w;      w += (size_t)2 * BATCH * H_DIM * 2;       // 128 KB
    unsigned* bar = (unsigned*)w;     w += 256;

    float* out = (float*)d_out;
    float* hy  = out + (size_t)T_STEPS * BATCH * 2 * H_DIM;
    float* cy  = hy + 2 * BATCH * H_DIM;

    // ---- pack operands into bf16 WMMA fragment layout ----------------------
    {
        int n = ROWS * IN_DIM;
        pack_a_bf16<<<(n + 255) / 256, 256, 0, stream>>>(x, xfrag, ROWS, IN_DIM);
    }
    {
        int n = G_DIM * IN_DIM;
        pack_b_bf16<<<(n + 255) / 256, 256, 0, stream>>>(w_ih_f, wihfrag, G_DIM, IN_DIM);
        pack_b_bf16<<<(n + 255) / 256, 256, 0, stream>>>(w_ih_r, wihfrag + (size_t)G_DIM * IN_DIM, G_DIM, IN_DIM);
    }
    {
        int n = G_DIM * H_DIM;
        pack_b_bf16<<<(n + 255) / 256, 256, 0, stream>>>(w_hh_f, whhfrag, G_DIM, H_DIM);
        pack_b_bf16<<<(n + 255) / 256, 256, 0, stream>>>(w_hh_r, whhfrag + (size_t)G_DIM * H_DIM, G_DIM, H_DIM);
    }

    // ---- zero initial h fragments and barrier counter (every call) ---------
    {
        size_t nb = (size_t)2 * BATCH * H_DIM * 2 * 2 + 256;   // hfragA+hfragB+bar
        zero_bytes<<<(unsigned)((nb + 255) / 256), 256, 0, stream>>>((char*)hfragA, nb);
    }

    // ---- phase 1: input projection GEMM (both directions) -------------------
    gx_gemm<<<32000, 128, 0, stream>>>(xfrag, wihfrag,
                                       b_ih_f, b_hh_f, b_ih_r, b_hh_r, gx);

    // ---- phase 2: all 500 recurrence steps in one persistent kernel ---------
    lstm_persistent<<<STEP_BLOCKS, 128, 0, stream>>>(gx, whhfrag, hfragA, hfragB,
                                                     out, hy, cy, bar);
}
